// CrossAttention_2740189135600
// MI455X (gfx1250) — matrix operand order repaired and checked by reading, not compiled, not run on hardware
//
#include <hip/hip_runtime.h>
#include <cstdint>
#include <cstddef>

// Problem constants (from reference)
#define B_      4
#define N_      2048
#define M_      2048
#define DIM_    512
#define HEADS_  8
#define DH_     64
#define INNER_  512
#define JPAD_   2080          // keys padded: 1 null + 2048 ctx + 31 pad = 65*32
#define NEG_MASK (-1.0e30f)   // masked (row or col) entries
#define NEG_PAD  (-1.0e38f)   // padding columns beyond j=2048 (strictly more negative)

typedef __bf16 bf16_t;
typedef __attribute__((ext_vector_type(16))) __bf16         v16bf;
typedef __attribute__((ext_vector_type(8)))  float          v8f;
typedef __attribute__((ext_vector_type(8)))  unsigned short ushort8;

// ---------- scalar helpers ----------
static __device__ __forceinline__ unsigned short f2bf(float f) {
  union { float f; unsigned u; } v; v.f = f;
  unsigned r = v.u + 0x7FFFu + ((v.u >> 16) & 1u);   // round-to-nearest-even
  return (unsigned short)(r >> 16);
}
static __device__ __forceinline__ bf16_t bfbits(unsigned short u) {
  union { unsigned short u; bf16_t b; } v; v.u = u; return v.b;
}

// ---------- WMMA fragment loaders (gfx1250 16-bit layouts, CDNA5 ISA) ----------
// A 16x32 (MxK): lanes 0-15 rows 0-15 / lanes 16-31 rows 0-15; per lane the 16
// elements are K = kh*8 + {0..7} and K = 16 + kh*8 + {0..7}  (kh = lane>>4).
static __device__ __forceinline__ v16bf load_a16x32(const unsigned short* base, int ld,
                                                    int kk0, int lane) {
  int r = lane & 15, kh = lane >> 4;
  const ushort8* p0 = (const ushort8*)(base + (size_t)r * ld + kk0 + kh * 8);
  const ushort8* p1 = (const ushort8*)(base + (size_t)r * ld + kk0 + 16 + kh * 8);
  ushort8 lo = *p0, hi = *p1;
  v16bf a;
#pragma unroll
  for (int i = 0; i < 8; ++i) { a[i] = bfbits(lo[i]); a[i + 8] = bfbits(hi[i]); }
  return a;
}
// B 32x16 (KxN): lane col = lane&15; per lane K = kh*8 + {0..7} and K = 16 + kh*8 + {0..7}.
// `base` points at a [16 x ld] block stored with one row per output column.
static __device__ __forceinline__ v16bf load_b32x16(const unsigned short* base, int ld,
                                                    int kk0, int lane) {
  int c = lane & 15, kh = lane >> 4;
  const ushort8* p0 = (const ushort8*)(base + (size_t)c * ld + kk0 + kh * 8);
  const ushort8* p1 = (const ushort8*)(base + (size_t)c * ld + kk0 + 16 + kh * 8);
  ushort8 lo = *p0, hi = *p1;
  v16bf b;
#pragma unroll
  for (int i = 0; i < 8; ++i) { b[i] = bfbits(lo[i]); b[i + 8] = bfbits(hi[i]); }
  return b;
}
static __device__ __forceinline__ v8f wmma_bf16(v16bf a, v16bf b, v8f c) {
  return __builtin_amdgcn_wmma_f32_16x16x32_bf16(false, a, false, b, (short)0, c,
                                                 false, false);
}
// Full K=512 reduction for a 16x16 output tile: A rows row-major [*,512] bf16,
// Wt is [col][512] bf16 (pre-transposed weight).
static __device__ __forceinline__ v8f gemm_tile512(const unsigned short* Arow0,
                                                   const unsigned short* Wcol0, int lane) {
  v8f acc = {};
#pragma unroll
  for (int kk = 0; kk < 512; kk += 32) {
    v16bf a = load_a16x32(Arow0, 512, kk, lane);
    v16bf b = load_b32x16(Wcol0, 512, kk, lane);
    acc = wmma_bf16(a, b, acc);
  }
  return acc;
}

// ---------- conversion kernels ----------
__global__ void cvt_bf16_kernel(const float* __restrict__ src,
                                unsigned short* __restrict__ dst, int n) {
  for (int i = blockIdx.x * blockDim.x + threadIdx.x; i < n; i += gridDim.x * blockDim.x)
    dst[i] = f2bf(src[i]);
}
__global__ void cvt_tr_kernel(const float* __restrict__ W,
                              unsigned short* __restrict__ Wt, int K, int C) {
  int n = K * C;
  for (int i = blockIdx.x * blockDim.x + threadIdx.x; i < n; i += gridDim.x * blockDim.x) {
    int k = i / C, c = i % C;
    Wt[(size_t)c * K + k] = f2bf(W[i]);
  }
}

// ---------- Q projection: Q = tanh(x @ Wq), stored [B,H,N,64] bf16 ----------
__global__ __launch_bounds__(128) void proj_q_kernel(const unsigned short* __restrict__ xbf,
                                                     const unsigned short* __restrict__ WqT,
                                                     unsigned short* __restrict__ Qbf) {
  int lane = threadIdx.x & 31, wave = threadIdx.x >> 5;
  int r0 = blockIdx.x * 16;                         // 0..8191 rows (b*n flat)
  int c0 = (blockIdx.y * 4 + wave) * 16;            // 0..511 cols
  v8f acc = gemm_tile512(xbf + (size_t)r0 * DIM_, WqT + (size_t)c0 * DIM_, lane);
  int kh = lane >> 4, cl = lane & 15;
  int col = c0 + cl, h = col >> 6, d = col & 63;
#pragma unroll
  for (int e = 0; e < 8; ++e) {
    int rg = r0 + e + kh * 8;
    int b = rg >> 11, n = rg & 2047;
    Qbf[((size_t)((b * HEADS_ + h) * N_ + n)) * DH_ + d] = f2bf(tanhf(acc[e]));
  }
}

// ---------- KV projection: K = tanh(ctx@Wkv[:, :512]) -> [B,H,JPAD,64],
//                           V = ctx@Wkv[:, 512:]       -> transposed [B,H,64,JPAD]
__global__ __launch_bounds__(128) void proj_kv_kernel(const unsigned short* __restrict__ cbf,
                                                      const unsigned short* __restrict__ WkvT,
                                                      unsigned short* __restrict__ Kbf,
                                                      unsigned short* __restrict__ Vt) {
  int lane = threadIdx.x & 31, wave = threadIdx.x >> 5;
  int r0 = blockIdx.x * 16;                         // 0..8191 rows (b*m flat)
  int c0 = (blockIdx.y * 4 + wave) * 16;            // 0..1023 cols
  v8f acc = gemm_tile512(cbf + (size_t)r0 * DIM_, WkvT + (size_t)c0 * DIM_, lane);
  int kh = lane >> 4, cl = lane & 15;
  int col = c0 + cl;
#pragma unroll
  for (int e = 0; e < 8; ++e) {
    int rg = r0 + e + kh * 8;
    int b = rg >> 11, m = rg & 2047;
    if (col < 512) {
      int h = col >> 6, d = col & 63;
      Kbf[((size_t)(b * HEADS_ + h) * JPAD_ + (m + 1)) * DH_ + d] = f2bf(tanhf(acc[e]));
    } else {
      int c2 = col - 512, h = c2 >> 6, d = c2 & 63;
      Vt[((size_t)(b * HEADS_ + h) * DH_ + d) * JPAD_ + (m + 1)] = f2bf(acc[e]);
    }
  }
}

// ---------- null token + tail zeroing ----------
__global__ void kv_fixup_kernel(const float* __restrict__ null_key,
                                const float* __restrict__ null_value,
                                unsigned short* __restrict__ Kbf,
                                unsigned short* __restrict__ Vt) {
  int bh = blockIdx.x;                               // 32 = B*H
  size_t kb = (size_t)bh * JPAD_ * DH_;
  size_t vb = (size_t)bh * DH_ * JPAD_;
  int t = threadIdx.x;                               // 128
  if (t < DH_) {
    Kbf[kb + t] = f2bf(tanhf(null_key[t]));          // j = 0 null key (tanh'ed)
    Vt[vb + (size_t)t * JPAD_ + 0] = f2bf(null_value[t]);
  }
  const int tail = (JPAD_ - (M_ + 1)) * DH_;         // 31*64 pad elements
  for (int i = t; i < tail; i += blockDim.x) {
    int j = (M_ + 1) + i / DH_, d = i % DH_;
    Kbf[kb + (size_t)j * DH_ + d] = 0;
    Vt[vb + (size_t)d * JPAD_ + j] = 0;
  }
}

// ---------- padded column mask [B][JPAD] ----------
__global__ void cmask_kernel(const unsigned char* __restrict__ ctx_mask,
                             unsigned char* __restrict__ cmask) {
  int b = blockIdx.x;
  for (int j = threadIdx.x; j < JPAD_; j += blockDim.x) {
    unsigned char v;
    if (j == 0)           v = 1;                     // null token always attendable
    else if (j <= M_)     v = ctx_mask[(size_t)b * M_ + (j - 1)] ? 1 : 0;
    else                  v = 0;                     // tile padding
    cmask[(size_t)b * JPAD_ + j] = v;
  }
}

// ---------- fused flash attention: one 16-row Q tile per wave ----------
__global__ __launch_bounds__(128) void attn_kernel(const unsigned short* __restrict__ Qbf,
                                                   const unsigned short* __restrict__ Kbf,
                                                   const unsigned short* __restrict__ Vt,
                                                   const unsigned char* __restrict__ mask,
                                                   const unsigned char* __restrict__ cmask,
                                                   unsigned short* __restrict__ Obf) {
  __shared__ __align__(16) unsigned short pbuf[4][16 * 32];  // per-wave P staging
  int lane = threadIdx.x & 31, wave = threadIdx.x >> 5;
  int bh = blockIdx.y, b = bh >> 3, h = bh & 7;
  int r0 = blockIdx.x * 64 + wave * 16;              // q row within N
  int kh = lane >> 4, cl = lane & 15;
  const float scale = 0.125f;                        // 64^-0.5

  const unsigned short* Qrow = Qbf + ((size_t)bh * N_ + r0) * DH_;
  v16bf aq0 = load_a16x32(Qrow, DH_, 0,  lane);
  v16bf aq1 = load_a16x32(Qrow, DH_, 32, lane);

  bool rowOk[8];
#pragma unroll
  for (int e = 0; e < 8; ++e)
    rowOk[e] = mask[(size_t)b * N_ + r0 + e + kh * 8] != 0;

  float mrow[8], lrow[8];
  v8f o0 = {}, o1 = {}, o2 = {}, o3 = {};
#pragma unroll
  for (int e = 0; e < 8; ++e) { mrow[e] = -3.0e38f; lrow[e] = 0.0f; }

  const unsigned short* Kbase = Kbf + (size_t)bh * JPAD_ * DH_;
  const unsigned short* Vbase = Vt  + (size_t)bh * DH_ * JPAD_;
  const unsigned char*  cm    = cmask + (size_t)b * JPAD_;
  unsigned short* plds = pbuf[wave];

  for (int j0 = 0; j0 < JPAD_; j0 += 32) {
    // ---- scores: two 16-key halves, K-dim 64 = 2 WMMA steps each ----
    v8f sA = {}, sB = {};
    {
      const unsigned short* kb0 = Kbase + (size_t)j0 * DH_;
      sA = wmma_bf16(aq0, load_b32x16(kb0, DH_, 0,  lane), sA);
      sA = wmma_bf16(aq1, load_b32x16(kb0, DH_, 32, lane), sA);
      const unsigned short* kb1 = Kbase + (size_t)(j0 + 16) * DH_;
      sB = wmma_bf16(aq0, load_b32x16(kb1, DH_, 0,  lane), sB);
      sB = wmma_bf16(aq1, load_b32x16(kb1, DH_, 32, lane), sB);
    }
    int  jA = j0 + cl, jB = j0 + 16 + cl;
    bool cOkA = cm[jA] != 0, cOkB = cm[jB] != 0;
    float pa[8], pb[8];
#pragma unroll
    for (int e = 0; e < 8; ++e) {
      float va = sA[e] * scale, vb = sB[e] * scale;
      if (jA > M_)                 va = NEG_PAD;
      else if (!rowOk[e] || !cOkA) va = NEG_MASK;
      if (jB > M_)                 vb = NEG_PAD;
      else if (!rowOk[e] || !cOkB) vb = NEG_MASK;
      pa[e] = va; pb[e] = vb;
    }
    // ---- online softmax (rows live in 16-lane half-groups; xor<16 stays in-group)
    float alpha[8];
#pragma unroll
    for (int e = 0; e < 8; ++e) {
      float vm = fmaxf(pa[e], pb[e]);
#pragma unroll
      for (int msk = 1; msk < 16; msk <<= 1) vm = fmaxf(vm, __shfl_xor(vm, msk, 32));
      float mnew = fmaxf(mrow[e], vm);
      alpha[e] = __expf(mrow[e] - mnew);
      float ea = __expf(pa[e] - mnew), eb = __expf(pb[e] - mnew);
      float rs = ea + eb;
#pragma unroll
      for (int msk = 1; msk < 16; msk <<= 1) rs += __shfl_xor(rs, msk, 32);
      lrow[e] = lrow[e] * alpha[e] + rs;
      mrow[e] = mnew;
      pa[e] = ea; pb[e] = eb;
    }
#pragma unroll
    for (int e = 0; e < 8; ++e) {
      o0[e] *= alpha[e]; o1[e] *= alpha[e]; o2[e] *= alpha[e]; o3[e] *= alpha[e];
    }
    // ---- C-layout -> A-layout via wave-private LDS round trip ----
#pragma unroll
    for (int e = 0; e < 8; ++e) {
      int row = e + kh * 8;
      plds[row * 32 + cl]      = f2bf(pa[e]);
      plds[row * 32 + 16 + cl] = f2bf(pb[e]);
    }
    asm volatile("s_wait_dscnt 0" ::: "memory");
    v16bf pA = load_a16x32(plds, 32, 0, lane);
    // ---- O += P @ V  (V transposed: two 16B runs per lane per dh-slice) ----
    o0 = wmma_bf16(pA, load_b32x16(Vbase + (size_t)0 * 16 * JPAD_, JPAD_, j0, lane), o0);
    o1 = wmma_bf16(pA, load_b32x16(Vbase + (size_t)1 * 16 * JPAD_, JPAD_, j0, lane), o1);
    o2 = wmma_bf16(pA, load_b32x16(Vbase + (size_t)2 * 16 * JPAD_, JPAD_, j0, lane), o2);
    o3 = wmma_bf16(pA, load_b32x16(Vbase + (size_t)3 * 16 * JPAD_, JPAD_, j0, lane), o3);
  }
  // ---- normalize + store O as bf16 [B*N, 512] ----
#pragma unroll
  for (int e = 0; e < 8; ++e) {
    float inv = (lrow[e] > 0.0f) ? 1.0f / lrow[e] : 0.0f;
    size_t rg = (size_t)b * N_ + r0 + e + kh * 8;
    size_t base = rg * INNER_ + (size_t)h * DH_;
    Obf[base + 0 * 16 + cl] = f2bf(o0[e] * inv);
    Obf[base + 1 * 16 + cl] = f2bf(o1[e] * inv);
    Obf[base + 2 * 16 + cl] = f2bf(o2[e] * inv);
    Obf[base + 3 * 16 + cl] = f2bf(o3[e] * inv);
  }
}

// ---------- output projection: out = O @ Wo + bo (f32 out) ----------
__global__ __launch_bounds__(128) void proj_o_kernel(const unsigned short* __restrict__ Obf,
                                                     const unsigned short* __restrict__ WoT,
                                                     const float* __restrict__ bo,
                                                     float* __restrict__ out) {
  int lane = threadIdx.x & 31, wave = threadIdx.x >> 5;
  int r0 = blockIdx.x * 16;
  int c0 = (blockIdx.y * 4 + wave) * 16;
  v8f acc = gemm_tile512(Obf + (size_t)r0 * INNER_, WoT + (size_t)c0 * INNER_, lane);
  int kh = lane >> 4, cl = lane & 15;
  int col = c0 + cl;
  float bias = bo[col];
#pragma unroll
  for (int e = 0; e < 8; ++e) {
    int rg = r0 + e + kh * 8;
    out[(size_t)rg * DIM_ + col] = acc[e] + bias;
  }
}

extern "C" void kernel_launch(void* const* d_in, const int* in_sizes, int n_in,
                              void* d_out, int out_size, void* d_ws, size_t ws_size,
                              hipStream_t stream) {
  (void)in_sizes; (void)n_in; (void)out_size; (void)ws_size;
  const float*         x     = (const float*)d_in[0];
  const float*         ctx   = (const float*)d_in[1];
  const unsigned char* maskp = (const unsigned char*)d_in[2];  // bool
  const unsigned char* cmsk  = (const unsigned char*)d_in[3];  // bool
  const float*         Wq    = (const float*)d_in[4];
  const float*         Wkv   = (const float*)d_in[5];
  const float*         Wo    = (const float*)d_in[6];
  const float*         bo    = (const float*)d_in[7];
  const float*         nk    = (const float*)d_in[8];
  const float*         nv    = (const float*)d_in[9];
  float* out = (float*)d_out;

  unsigned short* ws = (unsigned short*)d_ws;
  size_t off = 0;
  unsigned short* xbf  = ws + off; off += (size_t)B_ * N_ * DIM_;            // 4.19M
  unsigned short* cbf  = ws + off; off += (size_t)B_ * M_ * DIM_;            // 4.19M
  unsigned short* Qbf  = ws + off; off += (size_t)B_ * HEADS_ * N_ * DH_;    // 4.19M
  unsigned short* Kbf  = ws + off; off += (size_t)B_ * HEADS_ * JPAD_ * DH_; // 4.26M
  unsigned short* Vt   = ws + off; off += (size_t)B_ * HEADS_ * DH_ * JPAD_; // 4.26M
  unsigned short* WqT  = ws + off; off += (size_t)DIM_ * INNER_;
  unsigned short* WkvT = ws + off; off += (size_t)DIM_ * 2 * INNER_;
  unsigned short* WoT  = ws + off; off += (size_t)INNER_ * DIM_;
  unsigned char*  cmp  = (unsigned char*)(ws + off);                         // B*JPAD bytes
  unsigned short* Obf  = cbf;  // ctx bf16 is dead after proj_kv -> reuse for O

  cvt_bf16_kernel<<<2048, 256, 0, stream>>>(x,   xbf, B_ * N_ * DIM_);
  cvt_bf16_kernel<<<2048, 256, 0, stream>>>(ctx, cbf, B_ * M_ * DIM_);
  cvt_tr_kernel<<<512,  256, 0, stream>>>(Wq,  WqT,  DIM_,  INNER_);
  cvt_tr_kernel<<<1024, 256, 0, stream>>>(Wkv, WkvT, DIM_,  2 * INNER_);
  cvt_tr_kernel<<<512,  256, 0, stream>>>(Wo,  WoT,  INNER_, DIM_);
  cmask_kernel<<<B_, 256, 0, stream>>>(cmsk, cmp);

  dim3 gq(512, 8);   proj_q_kernel <<<gq,  128, 0, stream>>>(xbf, WqT, Qbf);
  dim3 gkv(512, 16); proj_kv_kernel<<<gkv, 128, 0, stream>>>(cbf, WkvT, Kbf, Vt);
  kv_fixup_kernel<<<B_ * HEADS_, 128, 0, stream>>>(nk, nv, Kbf, Vt);
  dim3 ga(N_ / 64, B_ * HEADS_);
  attn_kernel<<<ga, 128, 0, stream>>>(Qbf, Kbf, Vt, maskp, cmp, Obf);
  dim3 go(512, 8);   proj_o_kernel <<<go,  128, 0, stream>>>(Obf, WoT, bo, out);
}
